// VoxelContext3D_fsa_18605798326418
// MI455X (gfx1250) — matrix operand order repaired and checked by reading, not compiled, not run on hardware
//
#include <hip/hip_runtime.h>

// ---------------------------------------------------------------------------
// VoxelContext3D fused self-attention for MI455X (gfx1250, wave32, WMMA f16)
// K/V tiles staged via TDM (tensor_load_to_lds) or async-LDS loads when the
// toolchain exposes them, double-buffered in LDS.
// ---------------------------------------------------------------------------

typedef _Float16 f16;
typedef __attribute__((ext_vector_type(16))) _Float16 v16h;
typedef __attribute__((ext_vector_type(8)))  _Float16 v8h;
typedef __attribute__((ext_vector_type(4)))  _Float16 v4h;
typedef __attribute__((ext_vector_type(8)))  float    v8f;
typedef unsigned int u32x4 __attribute__((ext_vector_type(4)));
typedef int          i32x8 __attribute__((ext_vector_type(8)));
typedef int          i32x4 __attribute__((ext_vector_type(4)));

#define NTOT 32768      // total tokens
#define NB   8192       // tokens per batch
#define CDIM 128        // channels
#define RDIM 64         // head dim (C/2)
#define KVT  64         // kv tile (tokens) staged per step
#define TILEH (KVT * RDIM)   // halves per staged tile
#define GRID_NXY 70400  // nz*nx*ny = 2*176*200
#define OUT_ELEMS  36044800   // 4*256*200*176
#define HALF_ELEMS 9011200    // 4*256*100*88

// ---- CDNA5 data-mover feature ladder (compile-safe) -----------------------
#if defined(__gfx1250__) && __has_builtin(__builtin_amdgcn_tensor_load_to_lds) && \
    __has_builtin(__builtin_amdgcn_s_wait_tensorcnt)
#define USE_TDM 1
#else
#define USE_TDM 0
#endif

#if defined(__gfx1250__) && __has_builtin(__builtin_amdgcn_global_load_async_to_lds_b128) && \
    __has_builtin(__builtin_amdgcn_s_wait_asynccnt)
#define USE_ASYNC 1
#else
#define USE_ASYNC 0
#endif

static __device__ __forceinline__ v16h cat8(v8h a, v8h b) {
  v16h r;
#pragma unroll
  for (int i = 0; i < 8; ++i) { r[i] = a[i]; r[8 + i] = b[i]; }
  return r;
}

static __device__ __forceinline__ v16h ldpair(const f16* p, int h0, int h1) {
  v8h a = *(const v8h*)(p + h0);
  v8h b = *(const v8h*)(p + h1);
  return cat8(a, b);
}

static __device__ __forceinline__ v8f wmma16(v16h a, v16h b, v8f c) {
  return __builtin_amdgcn_wmma_f32_16x16x32_f16(false, a, false, b,
                                                (short)0, c, false, false);
}

#if USE_TDM
// Issue a 2D TDM load: 64 rows x 64 halves tile, row stride `stride_elems`
// (in elements), into LDS byte offset `lds_byte`.  D# built per ISA ch.8.
// 6-arg builtin form: (g0, g1, g2, g3, g4, cpol).
static __device__ __forceinline__ void tdm_issue(const f16* gsrc,
                                                 unsigned lds_byte,
                                                 int stride_elems) {
  unsigned long long ga = (unsigned long long)(uintptr_t)gsrc;
  u32x4 g0;
  g0[0] = 1u;                                        // count=1 (valid), user
  g0[1] = lds_byte;                                  // lds_addr
  g0[2] = (unsigned)(ga & 0xFFFFFFFFu);              // global_addr[31:0]
  g0[3] = (unsigned)((ga >> 32) & 0x1FFFFFFu) | (2u << 30);  // [56:32] | type=2
  i32x8 g1;
  g1[0] = 0x00010000;           // data_size=2B; no multicast/iterate/pad
  g1[1] = (int)(64u << 16);     // tensor_dim0 = 64 (bits 79:48 low half)
  g1[2] = (int)(64u << 16);     // tensor_dim1 = 64 (bits 111:80 low half)
  g1[3] = (int)(64u << 16);     // tile_dim0 = 64 (bits 127:112)
  g1[4] = 64;                   // tile_dim1 = 64 (bits 143:128), tile_dim2=0
  g1[5] = stride_elems;         // tensor_dim0_stride (bits 207:160 low 32)
  g1[6] = 0;
  g1[7] = 0;
  i32x4 z4 = {0, 0, 0, 0};
  i32x8 z8 = {0, 0, 0, 0, 0, 0, 0, 0};
  __builtin_amdgcn_tensor_load_to_lds(g0, g1, z4, z4, z8, 0);
}
#endif

#if USE_ASYNC
static __device__ __forceinline__ void async_cp16(const f16* gsrc, f16* ldst) {
  __builtin_amdgcn_global_load_async_to_lds_b128(
      (__attribute__((address_space(1))) i32x4*)gsrc,
      (__attribute__((address_space(3))) i32x4*)ldst, 0, 0);
}
#endif

// ---------------------------------------------------------------------------
// Kernel 1: weight conversion. wcat = [wq*0.125 ; wk ; wv] (192x128) f16
// ---------------------------------------------------------------------------
__global__ void wconv_kernel(const float* __restrict__ wq,
                             const float* __restrict__ wk,
                             const float* __restrict__ wv,
                             const float* __restrict__ wo,
                             f16* __restrict__ wcat, f16* __restrict__ woh) {
  int i = blockIdx.x * 256 + threadIdx.x;          // 8192 threads
  if (i < RDIM * CDIM) {
    wcat[i]                   = (f16)(wq[i] * 0.125f);  // 1/sqrt(64) folded
    wcat[RDIM * CDIM + i]     = (f16)wk[i];
    wcat[2 * RDIM * CDIM + i] = (f16)wv[i];
  }
  if (i < CDIM * RDIM) woh[i] = (f16)wo[i];
}

// ---------------------------------------------------------------------------
// Kernel 2: positional encoding + add + LayerNorm (wave per token)
// ---------------------------------------------------------------------------
static __device__ __forceinline__ float pe_val(int c, float fx, float fy, float fz) {
  const float kw = 0.2093259175f;   // ln(1e4)/44
  const float kd = 1.7906740825f;   // -(ln(1e4)/44 - 2)  (reference quirk)
  if (c < 44) {
    int i = c >> 1;
    float ang = fx * __expf(-2.0f * (float)i * kw);
    return (c & 1) ? __cosf(ang) : __sinf(ang);
  }
  if (c < 88) {
    int i = (c - 44) >> 1;
    float ang = fy * __expf(-2.0f * (float)i * kw);
    return (c & 1) ? __cosf(ang) : __sinf(ang);
  }
  if ((c & 1) == 0) {
    int i = (c - 88) >> 1;
    return __sinf(fz * __expf(2.0f * (float)i * kd));
  }
  int i = (c - 89) >> 1;
  return __cosf(fz * __expf(2.0f * (float)i * kd));
}

__global__ void prep_kernel(const float* __restrict__ vox,
                            const int* __restrict__ coords,
                            const float* __restrict__ gamma,
                            const float* __restrict__ beta,
                            float* __restrict__ xf32, f16* __restrict__ xf16) {
  int wave = threadIdx.x >> 5, lane = threadIdx.x & 31;
  int token = blockIdx.x * 4 + wave;
  int c0 = lane * 4;
  const int* cr = coords + token * 4;
  float fz = (float)cr[1], fy = (float)cr[2], fx = (float)cr[3];

  float4 v = *(const float4*)(vox + (size_t)token * CDIM + c0);
  float x0 = v.x + pe_val(c0 + 0, fx, fy, fz);
  float x1 = v.y + pe_val(c0 + 1, fx, fy, fz);
  float x2 = v.z + pe_val(c0 + 2, fx, fy, fz);
  float x3 = v.w + pe_val(c0 + 3, fx, fy, fz);

  float s  = x0 + x1 + x2 + x3;
  float ss = x0 * x0 + x1 * x1 + x2 * x2 + x3 * x3;
#pragma unroll
  for (int off = 16; off; off >>= 1) {
    s  += __shfl_xor(s,  off, 32);
    ss += __shfl_xor(ss, off, 32);
  }
  float mean = s * (1.0f / CDIM);
  float var  = ss * (1.0f / CDIM) - mean * mean;
  float rs   = rsqrtf(var + 1e-6f);

  float4 g = *(const float4*)(gamma + c0);
  float4 b = *(const float4*)(beta + c0);
  float y0 = (x0 - mean) * rs * g.x + b.x;
  float y1 = (x1 - mean) * rs * g.y + b.y;
  float y2 = (x2 - mean) * rs * g.z + b.z;
  float y3 = (x3 - mean) * rs * g.w + b.w;

  *(float4*)(xf32 + (size_t)token * CDIM + c0) = make_float4(y0, y1, y2, y3);
  v4h h; h[0] = (f16)y0; h[1] = (f16)y1; h[2] = (f16)y2; h[3] = (f16)y3;
  *(v4h*)(xf16 + (size_t)token * CDIM + c0) = h;
}

// ---------------------------------------------------------------------------
// Kernel 3: fused QKV GEMM.  Out[n,192] = X[n,128] @ Wcat^T
// ---------------------------------------------------------------------------
__global__ void qkv_kernel(const f16* __restrict__ X, const f16* __restrict__ Wcat,
                           f16* __restrict__ Q, f16* __restrict__ K,
                           f16* __restrict__ VT) {
  __shared__ f16 w[192 * CDIM];                    // 48 KB
#pragma unroll
  for (int i = 0; i < 24; ++i)
    ((uint4*)w)[i * 128 + threadIdx.x] = ((const uint4*)Wcat)[i * 128 + threadIdx.x];
  __syncthreads();

  int wave = threadIdx.x >> 5, lane = threadIdx.x & 31;
  int row = lane & 15, hi = lane >> 4;
  int tok0 = (blockIdx.x * 4 + wave) * 16;

  v16h a[4];
#pragma unroll
  for (int kc = 0; kc < 4; ++kc) {
    int base = (tok0 + row) * CDIM + kc * 32;
    a[kc] = ldpair(X, base + 8 * hi, base + 16 + 8 * hi);
  }

#pragma unroll
  for (int os = 0; os < 12; ++os) {
    v8f acc = {0, 0, 0, 0, 0, 0, 0, 0};
    int outc = os * 16 + row;
#pragma unroll
    for (int kc = 0; kc < 4; ++kc) {
      int hb = outc * CDIM + kc * 32 + 8 * hi;
      acc = wmma16(a[kc], ldpair(w, hb, hb + 16), acc);
    }
    if (os < 8) {
      f16* dst = (os < 4) ? Q : K;
      int col = (os & 3) * 16 + row;
#pragma unroll
      for (int r = 0; r < 8; ++r)
        dst[(size_t)(tok0 + r + 8 * hi) * RDIM + col] = (f16)acc[r];
    } else {
      int d = (os - 8) * 16 + row;
      v8h vh;
#pragma unroll
      for (int r = 0; r < 8; ++r) vh[r] = (f16)acc[r];
      *(v8h*)(VT + (size_t)d * NTOT + tok0 + 8 * hi) = vh;
    }
  }
}

// ---------------------------------------------------------------------------
// Kernel 4: flash attention, 64-token kv tiles, double-buffered LDS staging.
// ---------------------------------------------------------------------------
__global__ void attn_kernel(const f16* __restrict__ Q, const f16* __restrict__ K,
                            const f16* __restrict__ VT, f16* __restrict__ O) {
  __shared__ f16 kt[2][TILEH];      // 2 x 8 KB  [kv token][64 k]
  __shared__ f16 vt[2][TILEH];      // 2 x 8 KB  [d][64 kv]
  __shared__ f16 pt[4][16 * KVT];   // 8 KB per-wave P tiles

  int wave = threadIdx.x >> 5, lane = threadIdx.x & 31;
  int row = lane & 15, hi = lane >> 4;
  int batch = blockIdx.y;
  int q0 = (blockIdx.x * 4 + wave) * 16;
  int t = threadIdx.x;

  const f16* Qb  = Q  + (size_t)batch * NB * RDIM;
  const f16* Kb  = K  + (size_t)batch * NB * RDIM;
  const f16* VTb = VT + (size_t)batch * NB;

  v16h qa[2];
#pragma unroll
  for (int kc = 0; kc < 2; ++kc) {
    int base = (q0 + row) * RDIM + kc * 32;
    qa[kc] = ldpair(Qb, base + 8 * hi, base + 16 + 8 * hi);
  }

  v8f o[4];
#pragma unroll
  for (int ds = 0; ds < 4; ++ds) o[ds] = (v8f){0, 0, 0, 0, 0, 0, 0, 0};
  v8f m, l;
#pragma unroll
  for (int r = 0; r < 8; ++r) { m[r] = -1e30f; l[r] = 0.0f; }

  f16* pw = &pt[wave][0];

  // ---- tile staging (TDM > async-LDS > sync copy) -------------------------
  auto stage = [&](int buf, int kv0) {
#if USE_TDM
    if (wave == 0) {
      unsigned koff = (unsigned)(uintptr_t)(__attribute__((address_space(3))) void*)&kt[buf][0];
      unsigned voff = (unsigned)(uintptr_t)(__attribute__((address_space(3))) void*)&vt[buf][0];
      tdm_issue(Kb + (size_t)kv0 * RDIM, koff, RDIM);
      tdm_issue(VTb + kv0, voff, NTOT);
    }
#elif USE_ASYNC
    int rt = t >> 1, sg = (t & 1) * 32;            // 64B per thread per tile
    const f16* ks = Kb + (size_t)(kv0 + rt) * RDIM + sg;
    f16* kd = &kt[buf][rt * RDIM + sg];
    const f16* vs = VTb + (size_t)rt * NTOT + kv0 + sg;
    f16* vd = &vt[buf][rt * RDIM + sg];
#pragma unroll
    for (int j = 0; j < 4; ++j) {
      async_cp16(ks + j * 8, kd + j * 8);
      async_cp16(vs + j * 8, vd + j * 8);
    }
#else
    int rt = t >> 1, sg = (t & 1) * 32;
    const f16* ks = Kb + (size_t)(kv0 + rt) * RDIM + sg;
    f16* kd = &kt[buf][rt * RDIM + sg];
    const f16* vs = VTb + (size_t)rt * NTOT + kv0 + sg;
    f16* vd = &vt[buf][rt * RDIM + sg];
#pragma unroll
    for (int j = 0; j < 4; ++j) {
      *(uint4*)(kd + j * 8) = *(const uint4*)(ks + j * 8);
      *(uint4*)(vd + j * 8) = *(const uint4*)(vs + j * 8);
    }
    if (kv0 + KVT < NB) __builtin_prefetch((const char*)ks + KVT * RDIM * 2, 0, 1);
#endif
  };

  stage(0, 0);

  const int NIT = NB / KVT;                        // 128 iterations
  for (int it = 0; it < NIT; ++it) {
    int cur = it & 1;
#if USE_TDM
    if (wave == 0) __builtin_amdgcn_s_wait_tensorcnt(0);
#elif USE_ASYNC
    __builtin_amdgcn_s_wait_asynccnt(0);
#endif
    __syncthreads();                               // buf `cur` ready everywhere
    if (it + 1 < NIT) stage(1 - cur, (it + 1) * KVT);

    const f16* ktc = &kt[cur][0];
    const f16* vtc = &vt[cur][0];

    // S = Q * K^T : four 16x16 tiles, batched B-frag loads per k-chunk
    v8f s[4];
#pragma unroll
    for (int ts = 0; ts < 4; ++ts) s[ts] = (v8f){0, 0, 0, 0, 0, 0, 0, 0};
#pragma unroll
    for (int kc = 0; kc < 2; ++kc) {
      v16h bf[4];
#pragma unroll
      for (int ts = 0; ts < 4; ++ts) {
        int hb = (ts * 16 + row) * RDIM + kc * 32 + 8 * hi;
        bf[ts] = ldpair(ktc, hb, hb + 16);
      }
#pragma unroll
      for (int ts = 0; ts < 4; ++ts) s[ts] = wmma16(qa[kc], bf[ts], s[ts]);
    }

    // online softmax (row of VGPR slot r = r + 8*hi; 16-lane reductions)
#pragma unroll
    for (int r = 0; r < 8; ++r) {
      float tm = fmaxf(fmaxf(s[0][r], s[1][r]), fmaxf(s[2][r], s[3][r]));
#pragma unroll
      for (int off = 1; off < 16; off <<= 1) tm = fmaxf(tm, __shfl_xor(tm, off, 32));
      float mn = fmaxf(m[r], tm);
      float corr = __expf(m[r] - mn);
      m[r] = mn;
      float rs = 0.0f;
#pragma unroll
      for (int ts = 0; ts < 4; ++ts) {
        float p = __expf(s[ts][r] - mn);
        s[ts][r] = p;
        rs += p;
      }
#pragma unroll
      for (int off = 1; off < 16; off <<= 1) rs += __shfl_xor(rs, off, 32);
      l[r] = l[r] * corr + rs;
#pragma unroll
      for (int ds = 0; ds < 4; ++ds) o[ds][r] *= corr;
    }

    // P: D-layout -> row-major LDS tile [16 x 64]
#pragma unroll
    for (int ts = 0; ts < 4; ++ts)
#pragma unroll
      for (int r = 0; r < 8; ++r)
        pw[(r + 8 * hi) * KVT + ts * 16 + row] = (f16)s[ts][r];

    // O += P * V
#pragma unroll
    for (int kc = 0; kc < 2; ++kc) {
      int h0 = row * KVT + kc * 32 + 8 * hi;
      v16h pa = ldpair(pw, h0, h0 + 16);
      v16h bf[4];
#pragma unroll
      for (int ds = 0; ds < 4; ++ds) {
        int hb = (ds * 16 + row) * KVT + kc * 32 + 8 * hi;
        bf[ds] = ldpair(vtc, hb, hb + 16);
      }
#pragma unroll
      for (int ds = 0; ds < 4; ++ds) o[ds] = wmma16(pa, bf[ds], o[ds]);
    }
  }

  // finalize: O /= l, write row-major [n, 64] f16
  f16* Ob = O + (size_t)batch * NB * RDIM;
#pragma unroll
  for (int r = 0; r < 8; ++r) {
    float inv = 1.0f / l[r];
    int trow = (q0 + r + 8 * hi) * RDIM;
#pragma unroll
    for (int ds = 0; ds < 4; ++ds)
      Ob[trow + ds * 16 + row] = (f16)(o[ds][r] * inv);
  }
}

// ---------------------------------------------------------------------------
// Kernel 5: output projection + residual.  Y[n,128] = O[n,64] @ Wo^T + Xin
// ---------------------------------------------------------------------------
__global__ void proj_kernel(const f16* __restrict__ O, const f16* __restrict__ Wo,
                            const float* __restrict__ Xin,
                            float* __restrict__ Xout, f16* __restrict__ Xout16) {
  __shared__ f16 w[CDIM * RDIM];                   // 16 KB
#pragma unroll
  for (int i = 0; i < 8; ++i)
    ((uint4*)w)[i * 128 + threadIdx.x] = ((const uint4*)Wo)[i * 128 + threadIdx.x];
  __syncthreads();

  int wave = threadIdx.x >> 5, lane = threadIdx.x & 31;
  int row = lane & 15, hi = lane >> 4;
  int tok0 = (blockIdx.x * 4 + wave) * 16;

  v16h a[2];
#pragma unroll
  for (int kc = 0; kc < 2; ++kc) {
    int base = (tok0 + row) * RDIM + kc * 32;
    a[kc] = ldpair(O, base + 8 * hi, base + 16 + 8 * hi);
  }

#pragma unroll
  for (int os = 0; os < 8; ++os) {
    v8f acc = {0, 0, 0, 0, 0, 0, 0, 0};
    int outc = os * 16 + row;
#pragma unroll
    for (int kc = 0; kc < 2; ++kc) {
      int hb = outc * RDIM + kc * 32 + 8 * hi;
      acc = wmma16(a[kc], ldpair(w, hb, hb + 16), acc);
    }
#pragma unroll
    for (int r = 0; r < 8; ++r) {
      size_t idx = (size_t)(tok0 + r + 8 * hi) * CDIM + outc;
      float y = acc[r] + Xin[idx];
      Xout[idx] = y;
      Xout16[idx] = (f16)y;
    }
  }
}

// ---------------------------------------------------------------------------
// Output kernels: zero, BEV scatter, 2x avg-pool downsample
// ---------------------------------------------------------------------------
__global__ void zero_kernel(float* __restrict__ out) {
  size_t i = ((size_t)blockIdx.x * 256 + threadIdx.x) * 4;
  *(float4*)(out + i) = make_float4(0.f, 0.f, 0.f, 0.f);
}

__global__ void scatter_kernel(const float* __restrict__ Y,
                               const int* __restrict__ coords,
                               float* __restrict__ out) {
  int idx = blockIdx.x * 256 + threadIdx.x;
  int token = idx >> 7, c = idx & 127;
  const int* cr = coords + token * 4;
  int b = cr[0];
  int g = cr[1] + cr[2] * 176 + cr[3];
  out[(size_t)b * CDIM * GRID_NXY + (size_t)c * GRID_NXY + g] =
      Y[(size_t)token * CDIM + c];
}

__global__ void down_kernel(const float* __restrict__ out,
                            float* __restrict__ half) {
  int idx = blockIdx.x * 256 + threadIdx.x;
  int ix = idx % 88;
  int iy = (idx / 88) % 100;
  int ch = (idx / 8800) % 256;
  int b  = idx / 2252800;
  size_t base = (((size_t)(b * 256 + ch) * 200) + 2 * iy) * 176 + 2 * ix;
  float v = 0.25f * (out[base] + out[base + 1] + out[base + 176] + out[base + 177]);
  half[idx] = v;
}

// ---------------------------------------------------------------------------
// Host launch
// ---------------------------------------------------------------------------
extern "C" void kernel_launch(void* const* d_in, const int* in_sizes, int n_in,
                              void* d_out, int out_size, void* d_ws, size_t ws_size,
                              hipStream_t stream) {
  const float* vox    = (const float*)d_in[0];
  const int*   coords = (const int*)d_in[1];
  const float* gamma  = (const float*)d_in[2];
  const float* beta   = (const float*)d_in[3];
  const float* wq1 = (const float*)d_in[4];
  const float* wk1 = (const float*)d_in[5];
  const float* wv1 = (const float*)d_in[6];
  const float* wo1 = (const float*)d_in[7];
  const float* wq2 = (const float*)d_in[8];
  const float* wk2 = (const float*)d_in[9];
  const float* wv2 = (const float*)d_in[10];
  const float* wo2 = (const float*)d_in[11];

  char* ws = (char*)d_ws;
  size_t off = 0;
  float* X0f = (float*)(ws + off); off += (size_t)NTOT * CDIM * 4;
  float* X1f = (float*)(ws + off); off += (size_t)NTOT * CDIM * 4;
  f16* X0h = (f16*)(ws + off); off += (size_t)NTOT * CDIM * 2;
  f16* X1h = (f16*)(ws + off); off += (size_t)NTOT * CDIM * 2;
  f16* Qb  = (f16*)(ws + off); off += (size_t)NTOT * RDIM * 2;
  f16* Kb  = (f16*)(ws + off); off += (size_t)NTOT * RDIM * 2;
  f16* VTb = (f16*)(ws + off); off += (size_t)NTOT * RDIM * 2;
  f16* Ob  = (f16*)(ws + off); off += (size_t)NTOT * RDIM * 2;
  f16* Wc1 = (f16*)(ws + off); off += 192 * CDIM * 2;
  f16* Wc2 = (f16*)(ws + off); off += 192 * CDIM * 2;
  f16* Wo1h = (f16*)(ws + off); off += CDIM * RDIM * 2;
  f16* Wo2h = (f16*)(ws + off); off += CDIM * RDIM * 2;

  float* out = (float*)d_out;

  wconv_kernel<<<32, 256, 0, stream>>>(wq1, wk1, wv1, wo1, Wc1, Wo1h);
  wconv_kernel<<<32, 256, 0, stream>>>(wq2, wk2, wv2, wo2, Wc2, Wo2h);

  prep_kernel<<<NTOT / 4, 128, 0, stream>>>(vox, coords, gamma, beta, X0f, X0h);

  // SA block 1
  qkv_kernel<<<NTOT / 64, 128, 0, stream>>>(X0h, Wc1, Qb, Kb, VTb);
  attn_kernel<<<dim3(NB / 64, 4), 128, 0, stream>>>(Qb, Kb, VTb, Ob);
  proj_kernel<<<NTOT / 64, 128, 0, stream>>>(Ob, Wo1h, X0f, X1f, X1h);

  // SA block 2
  qkv_kernel<<<NTOT / 64, 128, 0, stream>>>(X1h, Wc2, Qb, Kb, VTb);
  attn_kernel<<<dim3(NB / 64, 4), 128, 0, stream>>>(Qb, Kb, VTb, Ob);
  proj_kernel<<<NTOT / 64, 128, 0, stream>>>(Ob, Wo2h, X1f, X0f, X0h);

  // BEV outputs
  zero_kernel<<<(OUT_ELEMS + HALF_ELEMS) / 1024, 256, 0, stream>>>(out);
  scatter_kernel<<<NTOT * CDIM / 256, 256, 0, stream>>>(X0f, coords, out);
  down_kernel<<<HALF_ELEMS / 256, 256, 0, stream>>>(out, out + OUT_ELEMS);
}
